// CPRLinear_64330020159908
// MI455X (gfx1250) — matrix operand order repaired and checked by reading, not compiled, not run on hardware
//
#include <hip/hip_runtime.h>

typedef __attribute__((ext_vector_type(16))) _Float16 v16h;
typedef __attribute__((ext_vector_type(8)))  _Float16 h8;
typedef __attribute__((ext_vector_type(8)))  float    v8f;
typedef __attribute__((ext_vector_type(4)))  int      i4;

static constexpr int BROWS  = 256;    // batch rows (M)
static constexpr int KDIM   = 8192;   // reduction dim (IN)
static constexpr int ODIM   = 8192;   // outputs (OUT)
static constexpr int KHIGH  = 2048;   // high-precision group width
static constexpr int KLOW   = 6144;   // low-precision group width
static constexpr int NTILE  = 64;     // output cols per workgroup
static constexpr int KCHUNK = 128;    // K per staged LDS chunk (== quant TILE)
static constexpr int LSTR   = 136;    // LDS row stride in halfs (128 + 8 pad)
static constexpr int NCHUNK = KDIM / KCHUNK;  // 64

// ---------------------------------------------------------------------------
// Kernel 1: x_perm[b][k] = (f16) x[b][col_indices[k]]
// ---------------------------------------------------------------------------
__global__ __launch_bounds__(256) void xperm_f16_kernel(
    const float* __restrict__ x, const int* __restrict__ ci,
    _Float16* __restrict__ xp)
{
  const int id = blockIdx.x * 256 + threadIdx.x;
  const int b  = id >> 13;           // / 8192
  const int k  = id & (KDIM - 1);
  xp[id] = (_Float16)x[(long)b * KDIM + ci[k]];
}

// ---------------------------------------------------------------------------
// Kernel 2: fused dequant + WMMA GEMM over a K-range.
// Grid: (ODIM/NTILE, nKSplits) blocks of 256 threads (8 waves).
// Block (bx,by) computes dst[by][0:256, n0:n0+64] over chunks [by*cPer, +cPer).
// Weights are streamed from HBM exactly once across the whole grid.
// ---------------------------------------------------------------------------
__global__ __launch_bounds__(256) void qgemm_wmma_kernel(
    const _Float16* __restrict__ xp,
    const int*  __restrict__ whq, const int*  __restrict__ wlq,
    const float* __restrict__ sh, const float* __restrict__ zh,
    const float* __restrict__ sl, const float* __restrict__ zl,
    const float* __restrict__ bias, float* __restrict__ dstBase,
    int cPer, int addBias)
{
  __shared__ _Float16 wtile[2][NTILE * LSTR];   // 2 x 17408 B

  const int tid  = threadIdx.x;
  const int wave = tid >> 5;
  const int lane = tid & 31;
  const int n0   = blockIdx.x * NTILE;
  const int c0   = blockIdx.y * cPer;
  const int cend = c0 + cPer;
  float* __restrict__ dst = dstBase + (size_t)blockIdx.y * BROWS * ODIM;

  // ---- cooperative dequant-staging role: 4 threads per output row ----
  const int srow  = tid >> 2;        // 0..63  local output row
  const int spart = tid & 3;         // 0..3   (32 k-values each)
  const int orow  = n0 + srow;

  int   qv[32];
  float scl, zp;

  auto wsrc = [&](int c) -> const int* {        // weight address for a chunk
    const int kg = c * KCHUNK + spart * 32;
    if (kg < KHIGH) return whq + (long)orow * KHIGH + kg;
    return wlq + (long)orow * KLOW + (kg - KHIGH);
  };

  auto load_chunk = [&](int c) {
    const int kg = c * KCHUNK + spart * 32;     // global K of first element
    const int* src;
    if (kg < KHIGH) {                           // uniform per chunk
      const int t = kg >> 7;
      scl = sh[t * ODIM + orow];
      zp  = zh[t * ODIM + orow];
      src = whq + (long)orow * KHIGH + kg;
    } else {
      const int col = kg - KHIGH;
      const int t   = col >> 7;
      scl = sl[t * ODIM + orow];
      zp  = zl[t * ODIM + orow];
      src = wlq + (long)orow * KLOW + col;
    }
    const i4* p = (const i4*)src;               // 8 x b128 loads (128 B/thread)
#pragma unroll
    for (int i = 0; i < 8; ++i) {
      i4 v = p[i];
      qv[4*i+0] = v.x; qv[4*i+1] = v.y; qv[4*i+2] = v.z; qv[4*i+3] = v.w;
    }
    // prime L2/WGP$ two chunks ahead (no VGPR dest, no counters)
    if (c + 2 < cend) {
      const char* pf = (const char*)wsrc(c + 2);
      __builtin_prefetch(pf, 0, 3);
      __builtin_prefetch(pf + 64, 0, 3);
    }
  };

  auto store_chunk = [&](int buf) {
    _Float16* dstL = &wtile[buf][srow * LSTR + spart * 32];
#pragma unroll
    for (int i = 0; i < 4; ++i) {
      h8 hv;
#pragma unroll
      for (int j = 0; j < 8; ++j)
        hv[j] = (_Float16)(((float)qv[8*i + j] - zp) * scl);
      *(h8*)(dstL + 8*i) = hv;                  // 16 B aligned LDS stores
    }
  };

  // ---- compute role: wave handles rows [32*wave, 32*wave+32), 4 N-tiles ----
  const int m0    = wave * 32;
  const int hrow  = lane & 15;       // M-row (A) / N-col (B) within a 16-tile
  const int half  = lane >> 4;       // half-wave
  const int aKoff = half * 8;        // A layout: lower half K+0..7/+16..23
  const int bKoff = half * 8;        // B layout: lower half K+0..7/+16..23

  v8f acc[2][4];
  const v8f vzero = {0.f,0.f,0.f,0.f,0.f,0.f,0.f,0.f};
#pragma unroll
  for (int i = 0; i < 2; ++i)
#pragma unroll
    for (int j = 0; j < 4; ++j) acc[i][j] = vzero;

  load_chunk(c0);
  store_chunk(0);
  __syncthreads();

  for (int c = c0; c < cend; ++c) {
    const int buf = c & 1;
    if (c + 1 < cend) {
      load_chunk(c + 1);                        // prefetch next weights -> regs
      // prime next chunk's A rows so late A-frag loads hit L0 (2 lines/row)
      const char* pa0 =
          (const char*)(xp + (long)(m0 + hrow) * KDIM + (c + 1) * KCHUNK);
      __builtin_prefetch(pa0, 0, 3);
      __builtin_prefetch(pa0 + 128, 0, 3);
      const char* pa1 = pa0 + (long)16 * KDIM * sizeof(_Float16);
      __builtin_prefetch(pa1, 0, 3);
      __builtin_prefetch(pa1 + 128, 0, 3);
    }

#pragma unroll
    for (int kk = 0; kk < 4; ++kk) {            // 4 WMMA K-steps per chunk
      const int kbase = c * KCHUNK + kk * 32;
      v16h a[2], bf[4];
#pragma unroll
      for (int mt = 0; mt < 2; ++mt) {          // A frags straight from global xp
        const _Float16* pa =
            xp + (long)(m0 + mt*16 + hrow) * KDIM + kbase + aKoff;
        h8 lo = *(const h8*)pa;                 // K +0..7
        h8 hi = *(const h8*)(pa + 16);          // K +16..23
        a[mt] = __builtin_shufflevector(lo, hi,
                  0,1,2,3,4,5,6,7,8,9,10,11,12,13,14,15);
      }
#pragma unroll
      for (int nt = 0; nt < 4; ++nt) {          // B frags from LDS (conflict-free)
        const _Float16* pb =
            &wtile[buf][(nt*16 + hrow) * LSTR + kk*32 + bKoff];
        h8 lo = *(const h8*)pb;                 // K +0..7
        h8 hi = *(const h8*)(pb + 16);          // K +16..23
        bf[nt] = __builtin_shufflevector(lo, hi,
                  0,1,2,3,4,5,6,7,8,9,10,11,12,13,14,15);
      }
#pragma unroll
      for (int mt = 0; mt < 2; ++mt)
#pragma unroll
        for (int nt = 0; nt < 4; ++nt)
          acc[mt][nt] = __builtin_amdgcn_wmma_f32_16x16x32_f16(
              false, a[mt], false, bf[nt], (short)0, acc[mt][nt], false, false);
    }

    if (c + 1 < cend) store_chunk(buf ^ 1);     // fill other buffer
    __syncthreads();
  }

  // ---- epilogue (C/D layout: lane->N, VGPR r + half*8 -> M) ----
  float bv[4];
#pragma unroll
  for (int nt = 0; nt < 4; ++nt) bv[nt] = addBias ? bias[n0 + nt*16 + hrow] : 0.f;
#pragma unroll
  for (int mt = 0; mt < 2; ++mt)
#pragma unroll
    for (int nt = 0; nt < 4; ++nt)
#pragma unroll
      for (int r = 0; r < 8; ++r) {
        const int m = m0 + mt*16 + half*8 + r;
        dst[(long)m * ODIM + n0 + nt*16 + hrow] = acc[mt][nt][r] + bv[nt];
      }
}

// ---------------------------------------------------------------------------
// Kernel 3: out = P0 + P1 + bias  (partials are L2-resident: 16 MB vs 192 MB L2)
// ---------------------------------------------------------------------------
__global__ __launch_bounds__(256) void reduce_kernel(
    const float* __restrict__ P, const float* __restrict__ bias,
    float* __restrict__ out)
{
  const int id = blockIdx.x * 256 + threadIdx.x;
  out[id] = P[id] + P[id + (size_t)BROWS * ODIM] + bias[id & (ODIM - 1)];
}

// ---------------------------------------------------------------------------
extern "C" void kernel_launch(void* const* d_in, const int* in_sizes, int n_in,
                              void* d_out, int out_size, void* d_ws, size_t ws_size,
                              hipStream_t stream) {
  const float* x    = (const float*)d_in[0];
  const int*   whq  = (const int*)  d_in[1];
  const int*   wlq  = (const int*)  d_in[2];
  const float* sh   = (const float*)d_in[3];
  const float* zh   = (const float*)d_in[4];
  const float* sl   = (const float*)d_in[5];
  const float* zl   = (const float*)d_in[6];
  const float* bias = (const float*)d_in[7];
  const int*   ci   = (const int*)  d_in[8];

  const size_t xpBytes = (size_t)BROWS * KDIM * sizeof(_Float16);   // 4 MB
  const size_t pBytes  = (size_t)2 * BROWS * ODIM * sizeof(float);  // 16 MB

  _Float16* xp  = (_Float16*)d_ws;
  float*    out = (float*)d_out;

  // Stage 1: permute + f32->f16 convert of x
  xperm_f16_kernel<<<(BROWS * KDIM) / 256, 256, 0, stream>>>(x, ci, xp);

  if (ws_size >= xpBytes + pBytes) {
    // Stage 2: 2-way K-split GEMM into partials (256 WGs -> 2x occupancy,
    // weights still read exactly once), then reduce + bias.
    float* P = (float*)((char*)d_ws + xpBytes);
    qgemm_wmma_kernel<<<dim3(ODIM / NTILE, 2), 256, 0, stream>>>(
        xp, whq, wlq, sh, zh, sl, zl, bias, P, NCHUNK / 2, /*addBias=*/0);
    reduce_kernel<<<(BROWS * ODIM) / 256, 256, 0, stream>>>(P, bias, out);
  } else {
    // Fallback: single-pass GEMM straight into out with bias.
    qgemm_wmma_kernel<<<dim3(ODIM / NTILE, 1), 256, 0, stream>>>(
        xp, whq, wlq, sh, zh, sl, zl, bias, out, NCHUNK, /*addBias=*/1);
  }
}
